// Decoder_68401649156853
// MI455X (gfx1250) — matrix operand order repaired and checked by reading, not compiled, not run on hardware
//
#include <hip/hip_runtime.h>

// ---------------- problem constants (from reference) ----------------
#define B_   32
#define H_   512
#define D_   512
#define MC_  512
#define MF_  512
#define S_   50
#define E_   512
#define LAT_ 64
#define T_   30
#define V_   32000
#define XW_  (E_ + MC_ + MF_ + LAT_)   // 1600
#define G3_  (3 * H_)                  // 1536

typedef __attribute__((ext_vector_type(16))) __bf16 v16bf;
typedef __attribute__((ext_vector_type(8)))  __bf16 v8bf;
typedef __attribute__((ext_vector_type(4)))  __bf16 v4bf;
typedef __attribute__((ext_vector_type(8)))  float  v8f;
typedef __attribute__((ext_vector_type(4)))  float  v4f;

// ---------------- helpers ----------------
__device__ __forceinline__ __bf16 f2bf(float f) {
    unsigned int u = __float_as_uint(f);
    u += 0x7FFFu + ((u >> 16) & 1u);        // round-to-nearest-even
    unsigned short s = (unsigned short)(u >> 16);
    __bf16 b;
    __builtin_memcpy(&b, &s, 2);
    return b;
}

__device__ __forceinline__ float sigmf(float x) {
    return 1.f / (1.f + __expf(-x));
}

// float -> bf16, 4 elements/thread with 128-bit loads (n % 4 == 0)
__global__ void f2bf4_kernel(const float* __restrict__ src, __bf16* __restrict__ dst, int n4) {
    int i = blockIdx.x * blockDim.x + threadIdx.x;
    if (i >= n4) return;
    v4f v = *(const v4f*)(src + (size_t)i * 4);
    v4bf o;
#pragma unroll
    for (int j = 0; j < 4; ++j) o[j] = f2bf(v[j]);
    *(v4bf*)(dst + (size_t)i * 4) = o;
}

// h init: copy fp32 hidden and produce bf16 mirror
__global__ void init_h_kernel(const float* __restrict__ hidden,
                              float* __restrict__ h, __bf16* __restrict__ h_bf, int n) {
    int i = blockIdx.x * blockDim.x + threadIdx.x;
    if (i < n) { float v = hidden[i]; h[i] = v; h_bf[i] = f2bf(v); }
}

// ---------------- WMMA GEMM: C[M,N] = A[M,K] @ W[N,K]^T + bias ----------------
// A, W bf16 row-major; C fp32 with row stride ldc. Each wave: 16(M) x 64(N) tile,
// software-pipelined (double-buffered fragments), unrolled x2 over K.
// Requirements: M % 16 == 0, N % 64 == 0, K % 64 == 0.
struct FragSet {
    v8bf a0, a1;            // A 16x32 fragment (two 8-elem runs per lane)
    v8bf w[4][2];           // four B fragments (K-contiguous per lane)
};

__device__ __forceinline__ void load_frags(FragSet& f, const __bf16* __restrict__ arow,
                                           const __bf16* __restrict__ w0,
                                           const __bf16* __restrict__ w1,
                                           const __bf16* __restrict__ w2,
                                           const __bf16* __restrict__ w3,
                                           int k0, int hf) {
    f.a0 = *(const v8bf*)(arow + k0 + hf * 16);
    f.a1 = *(const v8bf*)(arow + k0 + hf * 16 + 8);
    const int kb = k0 + hf * 16;
    f.w[0][0] = *(const v8bf*)(w0 + kb); f.w[0][1] = *(const v8bf*)(w0 + kb + 8);
    f.w[1][0] = *(const v8bf*)(w1 + kb); f.w[1][1] = *(const v8bf*)(w1 + kb + 8);
    f.w[2][0] = *(const v8bf*)(w2 + kb); f.w[2][1] = *(const v8bf*)(w2 + kb + 8);
    f.w[3][0] = *(const v8bf*)(w3 + kb); f.w[3][1] = *(const v8bf*)(w3 + kb + 8);
}

__device__ __forceinline__ v16bf cat16(v8bf lo, v8bf hi) {
    return __builtin_shufflevector(lo, hi, 0, 1, 2, 3, 4, 5, 6, 7,
                                           8, 9, 10, 11, 12, 13, 14, 15);
}

__device__ __forceinline__ void mma_set(const FragSet& f, v8f acc[4]) {
    v16bf av = cat16(f.a0, f.a1);
#pragma unroll
    for (int j = 0; j < 4; ++j) {
        v16bf wv = cat16(f.w[j][0], f.w[j][1]);
        acc[j] = __builtin_amdgcn_wmma_f32_16x16x32_bf16(false, av, false, wv,
                                                         (short)0, acc[j], false, false);
    }
}

__global__ void wmma_gemm_bf16(const __bf16* __restrict__ A, int lda,
                               const __bf16* __restrict__ W, int ldw,
                               const float* __restrict__ bias,
                               float* __restrict__ C, int ldc,
                               int M, int N, int K) {
    const int wid = (blockIdx.x * blockDim.x + threadIdx.x) >> 5;
    const int TM = M >> 4, TN4 = N >> 6;
    if (wid >= TM * TN4) return;               // wave-uniform exit (EXEC all-1s for WMMA)
    const int mt  = wid / TN4;
    const int n0  = (wid % TN4) << 6;
    const int lane = threadIdx.x & 31;
    const int hf  = lane >> 4;
    const int l16 = lane & 15;

    const __bf16* arow = A + (size_t)(mt * 16 + l16) * lda;
    const __bf16* w0 = W + (size_t)(n0 + l16) * ldw;
    const __bf16* w1 = w0 + (size_t)16 * ldw;
    const __bf16* w2 = w0 + (size_t)32 * ldw;
    const __bf16* w3 = w0 + (size_t)48 * ldw;

    v8f acc[4] = {{}, {}, {}, {}};

    FragSet f0, f1;
    load_frags(f0, arow, w0, w1, w2, w3, 0, hf);
    for (int k0 = 0; k0 < K; k0 += 64) {
        load_frags(f1, arow, w0, w1, w2, w3, k0 + 32, hf);   // in flight during mma(f0)
        mma_set(f0, acc);
        if (k0 + 64 < K)
            load_frags(f0, arow, w0, w1, w2, w3, k0 + 64, hf); // in flight during mma(f1)
        mma_set(f1, acc);
    }

    // C/D layout: lane (hf,l16) holds rows mbase..mbase+7 of its column
    const int mbase = mt * 16 + hf * 8;
#pragma unroll
    for (int j = 0; j < 4; ++j) {
        const int n = n0 + j * 16 + l16;
        const float bv = bias ? bias[n] : 0.f;
#pragma unroll
        for (int r = 0; r < 8; ++r)
            C[(size_t)(mbase + r) * ldc + n] = acc[j][r] + bv;
    }
}

// ---------------- additive attention: scores + softmax + context ----------------
// one block per batch row b. q:[B,D]  Kmat:[S,B,D]  Wo:[D]  mem:[S,B,M]  mask:[S,B]  ctx:[B,M]
__global__ void attn_ctx_kernel(const float* __restrict__ q,
                                const float* __restrict__ Kmat,
                                const float* __restrict__ Wo,
                                const float* __restrict__ bo,
                                const float* __restrict__ mem,
                                const int*   __restrict__ mask,
                                float* __restrict__ ctx,
                                int S, int B, int D, int M) {
    const int b = blockIdx.x;
    const int tid = threadIdx.x;
    const int lane = tid & 31, wave = tid >> 5;
    __shared__ float s_p[64];

    const float* qr = q + (size_t)b * D;
    for (int s = wave; s < S; s += 8) {
        const float* kr = Kmat + ((size_t)s * B + b) * D;
        float part = 0.f;
        for (int d = lane; d < D; d += 32)
            part += tanhf(qr[d] + kr[d]) * Wo[d];
        for (int off = 16; off; off >>= 1)
            part += __shfl_xor(part, off, 32);
        if (lane == 0)
            s_p[s] = (mask[s * B + b] != 0) ? (part + bo[0]) : -__builtin_inff();
    }
    __syncthreads();
    if (tid == 0) {                             // softmax over S (S=50, trivial)
        float mx = -__builtin_inff();
        for (int s = 0; s < S; ++s) mx = fmaxf(mx, s_p[s]);
        float sum = 0.f;
        for (int s = 0; s < S; ++s) { float e = __expf(s_p[s] - mx); s_p[s] = e; sum += e; }
        float inv = 1.f / sum;
        for (int s = 0; s < S; ++s) s_p[s] *= inv;
    }
    __syncthreads();
    for (int m = tid; m < M; m += blockDim.x) {
        float acc = 0.f;
        for (int s = 0; s < S; ++s)
            acc += s_p[s] * mem[((size_t)s * B + b) * M + m];
        ctx[(size_t)b * M + m] = acc;
    }
}

// ---------------- x = concat(emb[tok], ctx_c, ctx_f, latent) -> bf16 ----------------
__global__ void build_x_kernel(const int* __restrict__ target, int t,
                               const float* __restrict__ emb,
                               const float* __restrict__ ctx_c,
                               const float* __restrict__ ctx_f,
                               const float* __restrict__ latent,
                               __bf16* __restrict__ x) {
    const int b = blockIdx.x;
    const int tok = (t == 0) ? 1 : target[(t - 1) * B_ + b];
    for (int i = threadIdx.x; i < XW_; i += blockDim.x) {
        float v;
        if      (i < E_)             v = emb[(size_t)tok * E_ + i];
        else if (i < E_ + MC_)       v = ctx_c[b * MC_ + (i - E_)];
        else if (i < E_ + MC_ + MF_) v = ctx_f[b * MF_ + (i - E_ - MC_)];
        else                         v = latent[b * LAT_ + (i - E_ - MC_ - MF_)];
        x[(size_t)b * XW_ + i] = f2bf(v);
    }
}

// ---------------- GRU gate fusion ----------------
__global__ void gru_update_kernel(const float* __restrict__ gi,
                                  const float* __restrict__ gh,
                                  float* __restrict__ h,
                                  __bf16* __restrict__ h_bf,
                                  __bf16* __restrict__ hrelu_bf) {
    int idx = blockIdx.x * blockDim.x + threadIdx.x;
    if (idx >= B_ * H_) return;
    int b = idx >> 9, j = idx & (H_ - 1);
    const float* gib = gi + (size_t)b * G3_;
    const float* ghb = gh + (size_t)b * G3_;
    float r = sigmf(gib[j]          + ghb[j]);
    float z = sigmf(gib[H_ + j]     + ghb[H_ + j]);
    float n = tanhf(gib[2 * H_ + j] + r * ghb[2 * H_ + j]);
    float hv = h[idx];
    float hn = (1.f - z) * n + z * hv;
    h[idx]        = hn;
    h_bf[idx]     = f2bf(hn);
    hrelu_bf[idx] = f2bf(fmaxf(hn, 0.f));
}

// ---------------- host side ----------------
static inline char* carve(char*& p, size_t bytes) {
    char* r = p;
    p += (bytes + 255) & ~(size_t)255;
    return r;
}

extern "C" void kernel_launch(void* const* d_in, const int* in_sizes, int n_in,
                              void* d_out, int out_size, void* d_ws, size_t ws_size,
                              hipStream_t stream) {
    const float* c_memory = (const float*)d_in[0];
    const int*   c_pad    = (const int*)  d_in[1];
    const float* f_memory = (const float*)d_in[2];
    const int*   f_pad    = (const int*)  d_in[3];
    const float* hidden   = (const float*)d_in[4];
    // d_in[5] cf_attn: unused by reference math
    const float* latent   = (const float*)d_in[6];
    const int*   target   = (const int*)  d_in[7];
    const float* emb      = (const float*)d_in[8];
    const float* cW_h = (const float*)d_in[9],  *cb_h = (const float*)d_in[10];
    const float* cW_m = (const float*)d_in[11], *cb_m = (const float*)d_in[12];
    const float* cW_o = (const float*)d_in[13], *cb_o = (const float*)d_in[14];
    const float* fW_h = (const float*)d_in[15], *fb_h = (const float*)d_in[16];
    const float* fW_m = (const float*)d_in[17], *fb_m = (const float*)d_in[18];
    const float* fW_o = (const float*)d_in[19], *fb_o = (const float*)d_in[20];
    const float* W_ih = (const float*)d_in[21], *b_ih = (const float*)d_in[22];
    const float* W_hh = (const float*)d_in[23], *b_hh = (const float*)d_in[24];
    const float* W_proj = (const float*)d_in[25], *b_proj = (const float*)d_in[26];
    float* out = (float*)d_out;

    // ---- workspace carve (~52 MB) ----
    char* p = (char*)d_ws;
    __bf16* wp_bf   = (__bf16*)carve(p, (size_t)V_  * H_  * 2);
    __bf16* wih_bf  = (__bf16*)carve(p, (size_t)G3_ * XW_ * 2);
    __bf16* whh_bf  = (__bf16*)carve(p, (size_t)G3_ * H_  * 2);
    __bf16* cwh_bf  = (__bf16*)carve(p, (size_t)D_  * H_  * 2);
    __bf16* fwh_bf  = (__bf16*)carve(p, (size_t)D_  * H_  * 2);
    __bf16* cwm_bf  = (__bf16*)carve(p, (size_t)D_  * MC_ * 2);
    __bf16* fwm_bf  = (__bf16*)carve(p, (size_t)D_  * MF_ * 2);
    __bf16* cmem_bf = (__bf16*)carve(p, (size_t)S_ * B_ * MC_ * 2);
    __bf16* fmem_bf = (__bf16*)carve(p, (size_t)S_ * B_ * MF_ * 2);
    float*  Kc      = (float*) carve(p, (size_t)S_ * B_ * D_ * 4);
    float*  Kf      = (float*) carve(p, (size_t)S_ * B_ * D_ * 4);
    float*  h       = (float*) carve(p, (size_t)B_ * H_ * 4);
    __bf16* h_bf    = (__bf16*)carve(p, (size_t)B_ * H_ * 2);
    __bf16* hrelu   = (__bf16*)carve(p, (size_t)B_ * H_ * 2);
    float*  qc      = (float*) carve(p, (size_t)B_ * D_ * 4);
    float*  qf      = (float*) carve(p, (size_t)B_ * D_ * 4);
    float*  ctxc    = (float*) carve(p, (size_t)B_ * MC_ * 4);
    float*  ctxf    = (float*) carve(p, (size_t)B_ * MF_ * 4);
    __bf16* x_bf    = (__bf16*)carve(p, (size_t)B_ * XW_ * 2);
    float*  gi      = (float*) carve(p, (size_t)B_ * G3_ * 4);
    float*  gh      = (float*) carve(p, (size_t)B_ * G3_ * 4);

    auto conv = [&](const float* s, __bf16* d, int n) {
        int n4 = n / 4;
        f2bf4_kernel<<<(n4 + 255) / 256, 256, 0, stream>>>(s, d, n4);
    };
    auto gemm = [&](const __bf16* A, int lda, const __bf16* W, int ldw,
                    const float* bias, float* C, int ldc, int M, int N, int K) {
        int waves  = (M / 16) * (N / 64);
        int blocks = (waves * 32 + 255) / 256;
        wmma_gemm_bf16<<<blocks, 256, 0, stream>>>(A, lda, W, ldw, bias, C, ldc, M, N, K);
    };

    // ---- one-time (per call) conversions ----
    conv(W_proj, wp_bf,  V_ * H_);
    conv(W_ih,   wih_bf, G3_ * XW_);
    conv(W_hh,   whh_bf, G3_ * H_);
    conv(cW_h,   cwh_bf, D_ * H_);
    conv(fW_h,   fwh_bf, D_ * H_);
    conv(cW_m,   cwm_bf, D_ * MC_);
    conv(fW_m,   fwm_bf, D_ * MF_);
    conv(c_memory, cmem_bf, S_ * B_ * MC_);
    conv(f_memory, fmem_bf, S_ * B_ * MF_);
    init_h_kernel<<<(B_ * H_ + 255) / 256, 256, 0, stream>>>(hidden, h, h_bf, B_ * H_);

    // ---- hoisted: K[s,b,:] = memory[s,b,:] @ W_m^T + b_m (time-invariant) ----
    gemm(cmem_bf, MC_, cwm_bf, MC_, cb_m, Kc, D_, S_ * B_, D_, MC_);
    gemm(fmem_bf, MF_, fwm_bf, MF_, fb_m, Kf, D_, S_ * B_, D_, MF_);

    // ---- decode loop ----
    for (int t = 0; t < T_; ++t) {
        // q = h @ W_h^T + b_h
        gemm(h_bf, H_, cwh_bf, H_, cb_h, qc, D_, B_, D_, H_);
        gemm(h_bf, H_, fwh_bf, H_, fb_h, qf, D_, B_, D_, H_);
        // attention + context
        attn_ctx_kernel<<<B_, 256, 0, stream>>>(qc, Kc, cW_o, cb_o, c_memory, c_pad, ctxc, S_, B_, D_, MC_);
        attn_ctx_kernel<<<B_, 256, 0, stream>>>(qf, Kf, fW_o, fb_o, f_memory, f_pad, ctxf, S_, B_, D_, MF_);
        // x = [emb, ctx_c, ctx_f, latent] (bf16)
        build_x_kernel<<<B_, 256, 0, stream>>>(target, t, emb, ctxc, ctxf, latent, x_bf);
        // GRU gates
        gemm(x_bf, XW_, wih_bf, XW_, b_ih, gi, G3_, B_, G3_, XW_);
        gemm(h_bf, H_,  whh_bf, H_,  b_hh, gh, G3_, B_, G3_, H_);
        gru_update_kernel<<<(B_ * H_ + 255) / 256, 256, 0, stream>>>(gi, gh, h, h_bf, hrelu);
        // logits[:, t, :] = relu(h_new) @ W_proj^T + b_proj  (dominant GEMM)
        gemm(hrelu, H_, wp_bf, H_, b_proj, out + (size_t)t * V_, T_ * V_, B_, V_, H_);
    }
}
